// MeshEdgeBlock_57552561766960
// MI455X (gfx1250) — matrix operand order repaired and checked by reading, not compiled, not run on hardware
//
#include <hip/hip_runtime.h>
#include <hip/hip_bf16.h>

// ---------------------------------------------------------------------------
// MeshGraphNet edge block: gather -> Linear(384,512)+SiLU -> Linear(512,128)
//                          -> LayerNorm -> +residual
// bf16 WMMA (v_wmma_f32_16x16x32_bf16), fp32 accumulate, wave32 / gfx1250.
// M_TILE=64 edges/block; 4-deep rotating B prefetch ring; LDS rows padded by
// 16B so per-lane A-fragment ds_load_b128 addresses spread across banks
// (unpadded strides were = 0 mod 256B -> 16-way bank conflicts).
// ---------------------------------------------------------------------------

#define N_NODES 100000
#define D_NODE  128
#define D_EDGE  128
#define D_IN    384           // 2*D_NODE + D_EDGE
#define D_HID   512
#define KT1     (D_IN  / 32)  // 12 k-steps GEMM1
#define NT1     (D_HID / 16)  // 32 n-tiles GEMM1
#define KT2     (D_HID / 32)  // 16 k-steps GEMM2
#define NT2     (D_EDGE/ 16)  // 8  n-tiles GEMM2
#define M_TILE  64            // edges per workgroup (4 M-tiles of 16)
#define X_STRIDE 392          // bf16 per x row  (384 + 8 pad, 16B aligned)
#define H_STRIDE 520          // bf16 per h row  (512 + 8 pad, 16B aligned)
#define Y_STRIDE 132          // f32  per y row  (128 + 4 pad)
#define LN_EPS  1e-5f

typedef __attribute__((ext_vector_type(16))) __bf16 v16bf;
typedef __attribute__((ext_vector_type(8)))  __bf16 v8bf;
typedef __attribute__((ext_vector_type(8)))  float  v8f;

static __device__ __forceinline__ __bf16 f2bf(float f) {
    union { float f; unsigned u; } x; x.f = f;
    unsigned r = x.u + 0x7FFFu + ((x.u >> 16) & 1u);   // round-to-nearest-even
    unsigned short h = (unsigned short)(r >> 16);
    __bf16 b; __builtin_memcpy(&b, &h, 2); return b;
}

static __device__ __forceinline__ v16bf join16(v8bf lo, v8bf hi) {
    return __builtin_shufflevector(lo, hi, 0,1,2,3,4,5,6,7,8,9,10,11,12,13,14,15);
}

static __device__ __forceinline__ v16bf loadB(const __bf16* __restrict__ base,
                                              int frag_idx, int lane) {
    const v8bf* bp = (const v8bf*)(base + ((size_t)frag_idx * 32 + lane) * 16);
    return join16(bp[0], bp[1]);
}

// ---------------------------------------------------------------------------
// Weight pre-pack: fp32 row-major -> bf16 WMMA B-fragment order.
// Fragment (kt,nt): lane l holds column N = nt*16 + (l&15); halves j=0..15 hold
// K = kt*32 + (l>>4)*16 + j  (dense 32x16 16-bit B layout, contiguous per lane).
// Dest index = ((kt*NT + nt)*32 + lane)*16 + j  -> each lane = two 16B loads.
// ---------------------------------------------------------------------------
__global__ void pack_w1_kernel(const float* __restrict__ W1, __bf16* __restrict__ out) {
    int i = blockIdx.x * blockDim.x + threadIdx.x;
    if (i >= KT1 * NT1 * 32 * 16) return;
    int j    =  i        & 15;
    int lane = (i >> 4)  & 31;
    int nt   = (i >> 9)  & 31;   // NT1 = 32
    int kt   =  i >> 14;
    int kl   = (lane >> 4) * 16 + j;
    int nl   =  lane & 15;
    out[i] = f2bf(W1[(kt * 32 + kl) * D_HID + nt * 16 + nl]);
}

__global__ void pack_w2_kernel(const float* __restrict__ W2, __bf16* __restrict__ out) {
    int i = blockIdx.x * blockDim.x + threadIdx.x;
    if (i >= KT2 * NT2 * 32 * 16) return;
    int j    =  i        & 15;
    int lane = (i >> 4)  & 31;
    int nt   = (i >> 9)  & 7;    // NT2 = 8
    int kt   =  i >> 12;
    int kl   = (lane >> 4) * 16 + j;
    int nl   =  lane & 15;
    out[i] = f2bf(W2[(kt * 32 + kl) * D_EDGE + nt * 16 + nl]);
}

// ---------------------------------------------------------------------------
// Main fused kernel. Block = 256 threads = 8 wave32, 64 edges.
// Wave w owns M-tile (w & 3); two waves per M-tile split the N-tiles
// (nt = (w>>2) + 2*i), so A fragments are register-cached once per wave.
// ---------------------------------------------------------------------------
__global__ __launch_bounds__(256)
void edge_block_kernel(const float* __restrict__ s_feat,
                       const float* __restrict__ d_feat,
                       const float* __restrict__ e_feat,
                       const int*   __restrict__ s_idx,
                       const int*   __restrict__ d_idx,
                       const __bf16* __restrict__ w1p,
                       const float* __restrict__ b1,
                       const __bf16* __restrict__ w2p,
                       const float* __restrict__ b2,
                       const float* __restrict__ gamma,
                       const float* __restrict__ beta,
                       float* __restrict__ out,
                       int E) {
    __shared__ __align__(16) __bf16 h_lds[M_TILE * H_STRIDE];        // 65 KB
    __shared__ __align__(16) float  xy_lds[M_TILE * X_STRIDE / 2];   // 49 KB (x bf16 / y f32 union)
    __bf16* x_lds = (__bf16*)xy_lds;                                 // 64 x 392 bf16
    float*  y_lds = xy_lds;                                          // 64 x 132 f32 (33 KB)

    const int tid  = threadIdx.x;
    const int wave = tid >> 5;
    const int lane = tid & 31;
    const int e0   = blockIdx.x * M_TILE;

    // ---- Phase 1: gather [src | dst | edge] rows into LDS as bf16 ---------
    for (int q = tid; q < M_TILE * (D_IN / 4); q += 256) {
        int row = q / (D_IN / 4);
        int c4  = q % (D_IN / 4);
        int e   = e0 + row;
        float4 v = make_float4(0.f, 0.f, 0.f, 0.f);
        if (e < E) {
            if (c4 < 32) {
                v = *(const float4*)(s_feat + (size_t)s_idx[e] * D_NODE + c4 * 4);
            } else if (c4 < 64) {
                v = *(const float4*)(d_feat + (size_t)d_idx[e] * D_NODE + (c4 - 32) * 4);
            } else {
                v = *(const float4*)(e_feat + (size_t)e * D_EDGE + (c4 - 64) * 4);
            }
        }
        __bf16* dst = x_lds + row * X_STRIDE + c4 * 4;
        dst[0] = f2bf(v.x); dst[1] = f2bf(v.y); dst[2] = f2bf(v.z); dst[3] = f2bf(v.w);
    }
    __syncthreads();

    // ---- Phase 2: h = SiLU(x @ W1 + b1) -----------------------------------
    const int m    = wave & 3;        // fixed M tile per wave (0..3)
    const int nb   = wave >> 2;       // n tiles: nb + 2*i
    const int g    = lane >> 4;
    const int nl   = lane & 15;
    const int arow = m * 16 + nl;     // A row for this lane
    const int rb   = m * 16 + g * 8;  // C layout: VGPR r -> M = g*8 + r

    // cache all 12 A fragments in registers (reused across 16 N tiles)
    v16bf afrag[KT1];
#pragma unroll
    for (int kt = 0; kt < KT1; ++kt) {
        v8bf lo = *(const v8bf*)(x_lds + arow * X_STRIDE + kt * 32 + g * 16);
        v8bf hi = *(const v8bf*)(x_lds + arow * X_STRIDE + kt * 32 + g * 16 + 8);
        afrag[kt] = join16(lo, hi);
    }
#pragma unroll 1
    for (int i = 0; i < 16; ++i) {
        const int nt = nb + 2 * i;    // covers 0..31 across the two waves
        // 4-deep rotating prefetch ring for B fragments (L2 latency hiding)
        v16bf bb[4];
#pragma unroll
        for (int p = 0; p < 4; ++p)
            bb[p] = loadB(w1p, p * NT1 + nt, lane);
        v8f acc = {};
#pragma unroll
        for (int kt = 0; kt < KT1; ++kt) {
            acc = __builtin_amdgcn_wmma_f32_16x16x32_bf16(
                false, afrag[kt], false, bb[kt & 3], (short)0, acc, false, false);
            if (kt + 4 < KT1)
                bb[kt & 3] = loadB(w1p, (kt + 4) * NT1 + nt, lane);
        }
        const int col  = nt * 16 + nl;
        const float bv = b1[col];
#pragma unroll
        for (int r = 0; r < 8; ++r) {
            float v  = acc[r] + bv;
            float sv = v / (1.0f + __expf(-v));   // SiLU
            h_lds[(rb + r) * H_STRIDE + col] = f2bf(sv);
        }
    }
    __syncthreads();

    // ---- Phase 3: y = h @ W2 + b2 (y overwrites x region) -----------------
#pragma unroll 1
    for (int i = 0; i < 4; ++i) {
        const int nt = nb + 2 * i;    // {0,2,4,6} or {1,3,5,7}
        v16bf bb[4];
#pragma unroll
        for (int p = 0; p < 4; ++p)
            bb[p] = loadB(w2p, p * NT2 + nt, lane);
        v8f acc = {};
#pragma unroll
        for (int kt = 0; kt < KT2; ++kt) {
            v8bf alo = *(const v8bf*)(h_lds + arow * H_STRIDE + kt * 32 + g * 16);
            v8bf ahi = *(const v8bf*)(h_lds + arow * H_STRIDE + kt * 32 + g * 16 + 8);
            v16bf a  = join16(alo, ahi);
            acc = __builtin_amdgcn_wmma_f32_16x16x32_bf16(
                false, a, false, bb[kt & 3], (short)0, acc, false, false);
            if (kt + 4 < KT2)
                bb[kt & 3] = loadB(w2p, (kt + 4) * NT2 + nt, lane);
        }
        const int col  = nt * 16 + nl;
        const float bv = b2[col];
#pragma unroll
        for (int r = 0; r < 8; ++r)
            y_lds[(rb + r) * Y_STRIDE + col] = acc[r] + bv;
    }
    __syncthreads();

    // ---- Phase 4: LayerNorm + gamma/beta + residual -----------------------
    {
        const int row = tid >> 2;         // 64 rows, 4 threads per row
        const int sub = tid & 3;          // 32 cols each
        const float* yr = y_lds + row * Y_STRIDE + sub * 32;
        float vals[32];
        float sum = 0.f, sq = 0.f;
#pragma unroll
        for (int c = 0; c < 32; ++c) {
            float v = yr[c];
            vals[c] = v; sum += v; sq += v * v;
        }
        // reduce across the 4-thread group (lanes aligned to 4 inside wave32)
        for (int msk = 1; msk < 4; msk <<= 1) {
            sum += __shfl_xor(sum, msk, 32);
            sq  += __shfl_xor(sq,  msk, 32);
        }
        const float mean = sum * (1.0f / D_EDGE);
        const float var  = sq * (1.0f / D_EDGE) - mean * mean;
        const float rstd = rsqrtf(var + LN_EPS);
        const int e = e0 + row;
        if (e < E) {
            const size_t ob = (size_t)e * D_EDGE + sub * 32;
#pragma unroll
            for (int c4 = 0; c4 < 8; ++c4) {
                float4 res = *(const float4*)(e_feat + ob + c4 * 4);
                int col = sub * 32 + c4 * 4;
                float4 o;
                o.x = (vals[c4*4+0] - mean) * rstd * gamma[col+0] + beta[col+0] + res.x;
                o.y = (vals[c4*4+1] - mean) * rstd * gamma[col+1] + beta[col+1] + res.y;
                o.z = (vals[c4*4+2] - mean) * rstd * gamma[col+2] + beta[col+2] + res.z;
                o.w = (vals[c4*4+3] - mean) * rstd * gamma[col+3] + beta[col+3] + res.w;
                *(float4*)(out + ob + c4 * 4) = o;
            }
        }
    }
}

// ---------------------------------------------------------------------------
extern "C" void kernel_launch(void* const* d_in, const int* in_sizes, int n_in,
                              void* d_out, int out_size, void* d_ws, size_t ws_size,
                              hipStream_t stream) {
    const float* s_feat = (const float*)d_in[0];
    const float* d_feat = (const float*)d_in[1];
    const float* e_feat = (const float*)d_in[2];
    const int*   s_idx  = (const int*)  d_in[3];
    const int*   d_idx  = (const int*)  d_in[4];
    const float* W1     = (const float*)d_in[5];
    const float* b1     = (const float*)d_in[6];
    const float* W2     = (const float*)d_in[7];
    const float* b2     = (const float*)d_in[8];
    const float* gamma  = (const float*)d_in[9];
    const float* beta   = (const float*)d_in[10];
    float* out = (float*)d_out;
    const int E = in_sizes[3];                    // src_indices count = N_EDGES

    __bf16* w1p = (__bf16*)d_ws;                                     // 384*512*2 B
    __bf16* w2p = (__bf16*)((char*)d_ws + (size_t)D_IN * D_HID * 2); // 512*128*2 B

    {
        int n = KT1 * NT1 * 32 * 16;
        pack_w1_kernel<<<(n + 255) / 256, 256, 0, stream>>>(W1, w1p);
    }
    {
        int n = KT2 * NT2 * 32 * 16;
        pack_w2_kernel<<<(n + 255) / 256, 256, 0, stream>>>(W2, w2p);
    }
    {
        int nblocks = (E + M_TILE - 1) / M_TILE;
        edge_block_kernel<<<nblocks, 256, 0, stream>>>(
            s_feat, d_feat, e_feat, s_idx, d_idx,
            w1p, b1, w2p, b2, gamma, beta, out, E);
    }
}
